// SparseTemporal_76768245449526
// MI455X (gfx1250) — compile-verified, matrix operand order checked by reading, not run on hardware
//
#include <hip/hip_runtime.h>

// ---------------- problem constants (from reference) ----------------
#define B_      2
#define T_      4
#define NPTS    32768          // points per (b,t)
#define C_      128
#define C3_     384            // 3*C
#define HD_     32
#define NSEQ    (B_ * NPTS)    // 65536 sequences
#define SEQ_PER_WG 16
#define THREADS 128            // 4 waves of 32

#define FH_STRIDE   136        // f16 halves per row: 128 + 8 pad (16B aligned rows)
#define QKVH_STRIDE 392        // f16 halves per row: 384 + 8 pad (16B aligned rows)
#define OH_STRIDE   136

typedef float     __attribute__((ext_vector_type(4)))  v4f;
typedef float     __attribute__((ext_vector_type(8)))  v8f;
typedef _Float16  __attribute__((ext_vector_type(4)))  v4h;
typedef _Float16  __attribute__((ext_vector_type(8)))  v8h;
typedef _Float16  __attribute__((ext_vector_type(16))) v16h;

static __device__ __forceinline__ v16h cat16(v8h lo, v8h hi) {
    v16h r;
    #pragma unroll
    for (int i = 0; i < 8; ++i) { r[i] = lo[i]; r[8 + i] = hi[i]; }
    return r;
}

// ---- tiny prep kernel: fp32 weights -> f16, transposed (wT[col][k]) ----
__global__ __launch_bounds__(256)
void convert_weights_kernel(const float* __restrict__ w_qkv,
                            const float* __restrict__ w_out,
                            _Float16* __restrict__ wqkvT,   // [384][128]
                            _Float16* __restrict__ woutT)   // [128][128]
{
    int id = blockIdx.x * 256 + threadIdx.x;        // 0 .. 65535
    if (id < C_ * C3_) {
        int row = id / C3_, col = id % C3_;         // w_qkv[row][col]
        wqkvT[col * C_ + row] = (_Float16)w_qkv[id];
    } else {
        int id2 = id - C_ * C3_;                    // 0 .. 16383
        int row = id2 / C_, col = id2 % C_;         // w_out[row][col]
        woutT[col * C_ + row] = (_Float16)w_out[id2];
    }
}

__global__ __launch_bounds__(THREADS)
void SparseTemporal_fused_kernel(const float* __restrict__ features,
                                 const int*   __restrict__ indices,
                                 const float* __restrict__ temp_embed,
                                 const _Float16* __restrict__ wqkvT,
                                 const _Float16* __restrict__ woutT,
                                 float* __restrict__ feat_out,
                                 int*   __restrict__ idx_out)
{
    __shared__ _Float16 f16_lds[64 * FH_STRIDE];     // 17,408 B : f tile (row = t*16+sIdx)
    __shared__ _Float16 qkv_lds[64 * QKVH_STRIDE];   // 50,176 B : q|k|v (f16)
    __shared__ _Float16 o16_lds[16 * OH_STRIDE];     //  4,352 B : attention out (f16)
    __shared__ int      trow[64];                    // total ~72.2 KB -> 4 WGs/WGP

    const int tid   = threadIdx.x;
    const int lane  = tid & 31;
    const int wave  = tid >> 5;
    const int lhalf = lane >> 4;        // 0: lanes 0-15, 1: lanes 16-31
    const int ln    = lane & 15;
    const int s0    = blockIdx.x * SEQ_PER_WG;

    // ---- phase 0: per-row time index from indices[:,0] % T ----
    if (tid < 64) {
        int t    = tid >> 4;
        int sIdx = tid & 15;
        int s    = s0 + sIdx;
        int b_i  = s >> 15;
        int n_i  = s & (NPTS - 1);
        long p   = (long)(b_i * T_ + t) * NPTS + n_i;
        trow[tid] = indices[p * 3] % T_;
    }
    __syncthreads();

    // ---- phase 1: stage f = features + temp_embed[time] into LDS as f16 ----
    #pragma unroll
    for (int it = 0; it < 16; ++it) {
        int idx  = tid + it * THREADS;   // 0..2047
        int r    = idx >> 5;             // row 0..63
        int c4   = idx & 31;             // float4 column
        int t    = r >> 4;
        int sIdx = r & 15;
        int s    = s0 + sIdx;
        int b_i  = s >> 15;
        int n_i  = s & (NPTS - 1);
        long p   = (long)(b_i * T_ + t) * NPTS + n_i;
        const v4f* src = (const v4f*)(features + p * C_);
        const v4f* te  = (const v4f*)(temp_embed + (long)trow[r] * C_);
        v4f a = __builtin_nontemporal_load(&src[c4]);
        v4f o = a + te[c4];
        v4h h;
        h[0] = (_Float16)o.x; h[1] = (_Float16)o.y;
        h[2] = (_Float16)o.z; h[3] = (_Float16)o.w;
        *(v4h*)&f16_lds[r * FH_STRIDE + c4 * 4] = h;   // 8B store
    }
    __syncthreads();

    // A-fragment loader: 16x32 f16 tile, rows rbase..rbase+15, K-step ks (K = 32*ks..+31)
    // ISA layout: half0 lanes hold K 0-7 & 16-23, half1 lanes hold K 8-15 & 24-31.
    #define LOAD_AFRAG(rbase, ks) \
        cat16(*(const v8h*)&f16_lds[((rbase) + ln) * FH_STRIDE + 32*(ks) + lhalf*8], \
              *(const v8h*)&f16_lds[((rbase) + ln) * FH_STRIDE + 32*(ks) + 16 + lhalf*8])

    // B-fragment loader from transposed f16 weights: column n*16+ln, K-step ks
    #define LOAD_BFRAG(WT, n, ks) \
        cat16(*(const v8h*)((WT) + (size_t)((n)*16 + ln) * C_ + 32*(ks) + lhalf*8), \
              *(const v8h*)((WT) + (size_t)((n)*16 + ln) * C_ + 32*(ks) + 16 + lhalf*8))

    // ---- phase 2a: GEMM1 Q-tiles (n = 0..7): only last-token rows (M-tile 3) ----
    for (int n = wave; n < 8; n += 4) {
        v8f acc = {};
        #pragma unroll
        for (int ks = 0; ks < 4; ++ks) {
            v16h bf = LOAD_BFRAG(wqkvT, n, ks);
            v16h af = LOAD_AFRAG(48, ks);
            acc = __builtin_amdgcn_wmma_f32_16x16x32_f16(
                false, af, false, bf, (short)0, acc, false, false);
        }
        #pragma unroll
        for (int i = 0; i < 8; ++i) {
            int row = 48 + i + lhalf * 8;              // C/D: VGPR i -> M=i / M=i+8
            qkv_lds[row * QKVH_STRIDE + n * 16 + ln] = (_Float16)acc[i];
        }
    }

    // ---- phase 2b: GEMM1 K/V-tiles (n = 8..23): all 4 M-tiles ----
    for (int n = 8 + wave; n < 24; n += 4) {
        v8f acc0 = {}, acc1 = {}, acc2 = {}, acc3 = {};
        #pragma unroll
        for (int ks = 0; ks < 4; ++ks) {
            v16h bf = LOAD_BFRAG(wqkvT, n, ks);
            v16h a0 = LOAD_AFRAG( 0, ks);
            v16h a1 = LOAD_AFRAG(16, ks);
            v16h a2 = LOAD_AFRAG(32, ks);
            v16h a3 = LOAD_AFRAG(48, ks);
            acc0 = __builtin_amdgcn_wmma_f32_16x16x32_f16(false, a0, false, bf, (short)0, acc0, false, false);
            acc1 = __builtin_amdgcn_wmma_f32_16x16x32_f16(false, a1, false, bf, (short)0, acc1, false, false);
            acc2 = __builtin_amdgcn_wmma_f32_16x16x32_f16(false, a2, false, bf, (short)0, acc2, false, false);
            acc3 = __builtin_amdgcn_wmma_f32_16x16x32_f16(false, a3, false, bf, (short)0, acc3, false, false);
        }
        #pragma unroll
        for (int i = 0; i < 8; ++i) {
            const int rr = i + lhalf * 8;
            const int cc = n * 16 + ln;
            qkv_lds[( 0 + rr) * QKVH_STRIDE + cc] = (_Float16)acc0[i];
            qkv_lds[(16 + rr) * QKVH_STRIDE + cc] = (_Float16)acc1[i];
            qkv_lds[(32 + rr) * QKVH_STRIDE + cc] = (_Float16)acc2[i];
            qkv_lds[(48 + rr) * QKVH_STRIDE + cc] = (_Float16)acc3[i];
        }
    }
    __syncthreads();

    // ---- phase 3: attention (last token), one thread per (seq, head), fp32 accum ----
    if (tid < 64) {
        const int sIdx = tid & 15;
        const int h    = tid >> 4;
        const float scale = 0.17677669529663687f;      // 1/sqrt(32)
        const _Float16* qp = &qkv_lds[(48 + sIdx) * QKVH_STRIDE + h * HD_];
        const _Float16* k0 = &qkv_lds[( 0 + sIdx) * QKVH_STRIDE + C_ + h * HD_];
        const _Float16* k1 = &qkv_lds[(16 + sIdx) * QKVH_STRIDE + C_ + h * HD_];
        const _Float16* k2 = &qkv_lds[(32 + sIdx) * QKVH_STRIDE + C_ + h * HD_];
        const _Float16* k3 = &qkv_lds[(48 + sIdx) * QKVH_STRIDE + C_ + h * HD_];
        float d0 = 0.f, d1 = 0.f, d2 = 0.f, d3 = 0.f;
        #pragma unroll
        for (int j = 0; j < HD_; ++j) {
            float q = (float)qp[j];
            d0 += q * (float)k0[j]; d1 += q * (float)k1[j];
            d2 += q * (float)k2[j]; d3 += q * (float)k3[j];
        }
        float sc0 = d0 * scale, sc1 = d1 * scale, sc2 = d2 * scale, sc3 = d3 * scale;
        float mx  = fmaxf(fmaxf(sc0, sc1), fmaxf(sc2, sc3));
        sc0 = expf(sc0 - mx); sc1 = expf(sc1 - mx);
        sc2 = expf(sc2 - mx); sc3 = expf(sc3 - mx);
        const float inv = 1.0f / (sc0 + sc1 + sc2 + sc3);
        const _Float16* v0 = &qkv_lds[( 0 + sIdx) * QKVH_STRIDE + 2 * C_ + h * HD_];
        const _Float16* v1 = &qkv_lds[(16 + sIdx) * QKVH_STRIDE + 2 * C_ + h * HD_];
        const _Float16* v2 = &qkv_lds[(32 + sIdx) * QKVH_STRIDE + 2 * C_ + h * HD_];
        const _Float16* v3 = &qkv_lds[(48 + sIdx) * QKVH_STRIDE + 2 * C_ + h * HD_];
        #pragma unroll
        for (int j = 0; j < HD_; ++j) {
            float o = sc0 * (float)v0[j] + sc1 * (float)v1[j]
                    + sc2 * (float)v2[j] + sc3 * (float)v3[j];
            o16_lds[sIdx * OH_STRIDE + h * HD_ + j] = (_Float16)(o * inv);
        }
    }
    __syncthreads();

    // ---- phase 4: GEMM2 (16x128)@(128x128) -> feat_out ----
    #define LOAD_OFRAG(ks) \
        cat16(*(const v8h*)&o16_lds[ln * OH_STRIDE + 32*(ks) + lhalf*8], \
              *(const v8h*)&o16_lds[ln * OH_STRIDE + 32*(ks) + 16 + lhalf*8])

    for (int n = wave; n < 8; n += 4) {
        v8f acc = {};
        #pragma unroll
        for (int ks = 0; ks < 4; ++ks) {
            v16h bf = LOAD_BFRAG(woutT, n, ks);
            v16h af = LOAD_OFRAG(ks);
            acc = __builtin_amdgcn_wmma_f32_16x16x32_f16(
                false, af, false, bf, (short)0, acc, false, false);
        }
        #pragma unroll
        for (int i = 0; i < 8; ++i) {
            int  row = i + lhalf * 8;                  // sIdx within tile
            long s   = s0 + row;
            feat_out[s * C_ + n * 16 + ln] = acc[i];   // coalesced 64B per 16 lanes
        }
    }

    // ---- phase 5: idx_out gather (last token row, col0 // T) ----
    if (tid < 48) {
        int sIdx = tid / 3, j = tid % 3;
        int s   = s0 + sIdx;
        int b_i = s >> 15;
        int n_i = s & (NPTS - 1);
        long p  = (long)(b_i * T_ + (T_ - 1)) * NPTS + n_i;
        int v   = indices[p * 3 + j];
        idx_out[(long)s * 3 + j] = (j == 0) ? (v / T_) : v;
    }
}

extern "C" void kernel_launch(void* const* d_in, const int* in_sizes, int n_in,
                              void* d_out, int out_size, void* d_ws, size_t ws_size,
                              hipStream_t stream) {
    const float* features   = (const float*)d_in[0];
    const int*   indices    = (const int*)  d_in[1];
    const float* temp_embed = (const float*)d_in[2];
    const float* w_qkv      = (const float*)d_in[3];
    const float* w_out      = (const float*)d_in[4];
    // d_in[5] (b) and d_in[6] (t) are compile-time constants here.

    _Float16* wqkvT = (_Float16*)d_ws;                 //  98,304 B
    _Float16* woutT = wqkvT + (size_t)C3_ * C_;        // +32,768 B (131,072 total)

    // 65,536 elements total -> 256 blocks x 256 threads
    hipLaunchKernelGGL(convert_weights_kernel, dim3(256), dim3(256), 0, stream,
                       w_qkv, w_out, wqkvT, woutT);

    float* feat_out = (float*)d_out;
    int*   idx_out  = (int*)((float*)d_out + (long)NSEQ * C_);

    hipLaunchKernelGGL(SparseTemporal_fused_kernel,
                       dim3(NSEQ / SEQ_PER_WG), dim3(THREADS), 0, stream,
                       features, indices, temp_embed, wqkvT, woutT,
                       feat_out, idx_out);
}
